// VQ_11957188952130
// MI455X (gfx1250) — matrix operand order repaired and checked by reading, not compiled, not run on hardware
//
#include <hip/hip_runtime.h>

// ---------------------------------------------------------------------------
// VQ nearest-codebook search on MI455X (gfx1250, wave32, WMMA + TDM).
//
// inputs:  d_in[0] = inputs          [N=32, C=256, H=32, W=32] f32
//          d_in[1] = inputs_thermal  [N=32, C=256, H=32, W=32] f32
//          d_in[2] = dictionary      [D=1024, C=256]           f32
// outputs (flat, f32): emb0 | emb_pt0 | idx0 | emb1 | emb_pt1 | idx1
//
// ws: [0..1023] f32 dict row norms | [1024..] bf16 dictionary copy (512 KB)
// ---------------------------------------------------------------------------

typedef __attribute__((ext_vector_type(16))) __bf16       v16bf;
typedef __attribute__((ext_vector_type(8)))  float        v8f;
typedef __attribute__((ext_vector_type(4)))  float        v4f;
typedef __attribute__((ext_vector_type(4)))  unsigned int u32x4;
typedef __attribute__((ext_vector_type(8)))  int          i32x8;
typedef __attribute__((ext_vector_type(4)))  int          i32x4;

#define C_DIM      256
#define D_DIM      1024
#define HW         1024          // H*W
#define EMB_ELEMS  8388608       // N*C*H*W
#define IDX_ELEMS  32768         // N*H*W
#define LDS_ROWS   64            // dictionary rows staged per chunk
#define LDS_PITCH  272           // bf16/row: 256 data + 16 pad; 544 B = 17*32 B
#define N_CHUNKS   (D_DIM / LDS_ROWS)

// ---------------------------------------------------------------------------
// 1a) dictionary row norms: dnorm[d] = sum_c dict[d][c]^2
// ---------------------------------------------------------------------------
__global__ __launch_bounds__(256)
void vq_dictnorm_kernel(const float* __restrict__ dict, float* __restrict__ dnorm)
{
    const int r = blockIdx.x * 256 + threadIdx.x;        // grid = 4 blocks
    const float* row = dict + r * C_DIM;
    float s = 0.0f;
    #pragma unroll 8
    for (int c = 0; c < C_DIM; c += 4) {
        const v4f v = *(const v4f*)(row + c);
        s = __builtin_fmaf(v.x, v.x, s);
        s = __builtin_fmaf(v.y, v.y, s);
        s = __builtin_fmaf(v.z, v.z, s);
        s = __builtin_fmaf(v.w, v.w, s);
    }
    dnorm[r] = s;
}

// ---------------------------------------------------------------------------
// 1b) dictionary f32 -> bf16 image (unpadded [1024][256]) for TDM staging
// ---------------------------------------------------------------------------
__global__ __launch_bounds__(256)
void vq_dictbf16_kernel(const float* __restrict__ dict, __bf16* __restrict__ dictbf)
{
    const int i = blockIdx.x * 256 + threadIdx.x;        // grid = 256 blocks
    const v4f v = *(const v4f*)(dict + i * 4);
    __bf16* dst = dictbf + i * 4;
    dst[0] = (__bf16)v.x; dst[1] = (__bf16)v.y;
    dst[2] = (__bf16)v.z; dst[3] = (__bf16)v.w;
}

// ---------------------------------------------------------------------------
// TDM: DMA one 64-row x 256-col bf16 chunk into LDS with hardware padding:
// pad_interval=6 (every 128 DWORDs = one 512 B row), pad_amount=7 (8 DWORDs
// = 32 B) -> LDS pitch 544 B = LDS_PITCH bf16, matching the compute layout.
// ---------------------------------------------------------------------------
__device__ __forceinline__
void tdm_stage_chunk(const __bf16* src, unsigned lds_byte_addr)
{
    const unsigned long long ga = (unsigned long long)(const void*)src;
    u32x4 g0;
    g0[0] = 1u;                                          // count=1, user mode
    g0[1] = lds_byte_addr;                               // LDS dest (bytes)
    g0[2] = (unsigned)ga;                                // global_addr[31:0]
    g0[3] = (unsigned)((ga >> 32) & 0x01FFFFFFu)         // global_addr[56:32]
          | 0x80000000u;                                 // type=2 (image)
    i32x8 g1;
    g1[0] = (int)((1u << 16)                             // data_size = 2 B
                | (1u << 20)                             // pad_enable
                | (6u << 22)                             // pad_interval: 128 DW
                | (7u << 25));                           // pad_amount: 8 DW
    g1[1] = (int)(256u << 16);                           // tensor_dim0 = 256
    g1[2] = (int)((unsigned)LDS_ROWS << 16);             // tensor_dim1 = 64
    g1[3] = (int)(256u << 16);                           // tile_dim0   = 256
    g1[4] = LDS_ROWS;                                    // tile_dim1   = 64
    g1[5] = 256;                                         // tensor_dim0_stride
    g1[6] = 0;
    g1[7] = 0;
    const i32x4 gz4 = {0, 0, 0, 0};                      // 2D tensor: groups 2/3 unused
    const i32x8 gz8 = {0, 0, 0, 0, 0, 0, 0, 0};
    __builtin_amdgcn_tensor_load_to_lds(g0, g1, gz4, gz4, gz8, 0);
}

// ---------------------------------------------------------------------------
// 2) main search: per wave, TWO 16-row tiles of x_flat vs all 1024 codes.
//    Each LDS B-fragment feeds two WMMAs (halved ds traffic per WMMA).
//    Block = 256 threads = 8 waves => 256 rows of T per block.
//    grid.x = 2 * (32768/256) = 256 ; blockIdx.x>>7 selects the input.
// ---------------------------------------------------------------------------
__global__ __launch_bounds__(256)
void vq_argmin_kernel(const float* __restrict__ x0,
                      const float* __restrict__ x1,
                      const __bf16* __restrict__ dictbf,
                      const float* __restrict__ dnorm,
                      float* __restrict__ out)
{
    __shared__ __bf16 lds[2][LDS_ROWS * LDS_PITCH];      // 2 x 34,816 B

    const int tid  = threadIdx.x;
    const int lane = tid & 31;
    const int wave = tid >> 5;
    const int n15  = lane & 15;                          // N column / A row in tile
    const int kh   = lane >> 4;                          // half-wave selector

    const int inp = blockIdx.x >> 7;
    const int bt  = blockIdx.x & 127;
    const float* __restrict__ x = inp ? x1 : x0;
    float* __restrict__ idx_out =
        out + (inp ? (4 * EMB_ELEMS + IDX_ELEMS) : (2 * EMB_ELEMS));

    // ---- A fragments: rows t0..t0+31 of x_flat, all 256 channels, bf16 ----
    // x_flat[t][c] lives at x[n*C*HW + c*HW + p], t = n*1024 + p.
    const int t0   = bt * 256 + wave * 32;
    const int nimg = t0 >> 10;                           // same image for both tiles
    const float* __restrict__ xlaneA = x + nimg * (C_DIM * HW) + ((t0 + n15) & 1023);
    const float* __restrict__ xlaneB = x + nimg * (C_DIM * HW) + ((t0 + 16 + n15) & 1023);

    // 16-bit A layout (16x32), contiguous K order matching B: lane holds K {16kh + 0..15}.
    v16bf a0[8], a1[8];                                  // 128 VGPRs, loaded once
    #pragma unroll
    for (int kc = 0; kc < 8; ++kc) {
        const int cbase = kc * 32 + 16 * kh;
        #pragma unroll
        for (int j = 0; j < 16; ++j) {
            const int k = cbase + j;
            a0[kc][j] = (__bf16)xlaneA[k * HW];
            a1[kc][j] = (__bf16)xlaneB[k * HW];
        }
    }

    // running argmin of (||d||^2 - 2*dot); row m = r + 8*kh, code d = d0 + n15
    float best_s0[8], best_s1[8];
    int   best_i0[8], best_i1[8];
    #pragma unroll
    for (int r = 0; r < 8; ++r) {
        best_s0[r] = __builtin_inff(); best_i0[r] = 0;
        best_s1[r] = __builtin_inff(); best_i1[r] = 0;
    }

    // ---- double-buffered TDM pipeline: DMA chunk k+1 under compute of k ----
    const unsigned lds_b0 = (unsigned)(unsigned long long)&lds[0][0];
    const unsigned lds_b1 = (unsigned)(unsigned long long)&lds[1][0];
    if (wave == 0) {
        tdm_stage_chunk(dictbf, lds_b0);                 // prologue: chunk 0
        __builtin_amdgcn_s_wait_tensorcnt(0);
    }
    __syncthreads();

    for (int chunk = 0; chunk < N_CHUNKS; ++chunk) {
        const int cur = chunk & 1;
        if (wave == 0 && chunk + 1 < N_CHUNKS)           // EXEC-independent, 1 issue/wave
            tdm_stage_chunk(dictbf + (chunk + 1) * (LDS_ROWS * C_DIM),
                            cur ? lds_b0 : lds_b1);

        const __bf16* __restrict__ buf = &lds[cur][0];
        const int dbase = chunk * LDS_ROWS;

        #pragma unroll
        for (int dt = 0; dt < LDS_ROWS / 16; ++dt) {
            // 16-bit B (32x16), contiguous K order as A: lane holds N=n15, K {16kh + 0..15}
            // contiguous -> one aligned 32 B read per k-chunk (544 % 32 == 0).
            const __bf16* brow = buf + (dt * 16 + n15) * LDS_PITCH + 16 * kh;
            v8f acc0 = {}, acc1 = {};
            #pragma unroll
            for (int kc = 0; kc < 8; ++kc) {
                const v16bf b = *(const v16bf*)(brow + kc * 32);
                acc0 = __builtin_amdgcn_wmma_f32_16x16x32_bf16(
                           false, a0[kc], false, b, (short)0, acc0, false, false);
                acc1 = __builtin_amdgcn_wmma_f32_16x16x32_bf16(
                           false, a1[kc], false, b, (short)0, acc1, false, false);
            }
            const int   d   = dbase + dt * 16 + n15;
            const float nrm = dnorm[d];
            #pragma unroll
            for (int r = 0; r < 8; ++r) {
                const float s0 = __builtin_fmaf(-2.0f, acc0[r], nrm);
                if (s0 < best_s0[r] || (s0 == best_s0[r] && d < best_i0[r])) {
                    best_s0[r] = s0; best_i0[r] = d;
                }
                const float s1 = __builtin_fmaf(-2.0f, acc1[r], nrm);
                if (s1 < best_s1[r] || (s1 == best_s1[r] && d < best_i1[r])) {
                    best_s1[r] = s1; best_i1[r] = d;
                }
            }
        }

        if (wave == 0 && chunk + 1 < N_CHUNKS)
            __builtin_amdgcn_s_wait_tensorcnt(0);        // chunk k+1 landed
        __syncthreads();                                 // buffers swap safely
    }

    // ---- finish argmin across the 16 lanes sharing each row (wave32) ----
    #pragma unroll
    for (int r = 0; r < 8; ++r) {
        float s = best_s0[r];  int i = best_i0[r];
        float u = best_s1[r];  int j = best_i1[r];
        #pragma unroll
        for (int off = 8; off >= 1; off >>= 1) {
            const float os = __shfl_xor(s, off, 32);
            const int   oi = __shfl_xor(i, off, 32);
            if (os < s || (os == s && oi < i)) { s = os; i = oi; }
            const float ou = __shfl_xor(u, off, 32);
            const int   oj = __shfl_xor(j, off, 32);
            if (ou < u || (ou == u && oj < j)) { u = ou; j = oj; }
        }
        if (n15 == 0) {
            idx_out[t0 + r + 8 * kh]      = (float)i;    // tile 0 rows
            idx_out[t0 + 16 + r + 8 * kh] = (float)j;    // tile 1 rows
        }
    }
}

// ---------------------------------------------------------------------------
// 3) gather: embedded[n,c,h,w] = dict[idx[n,h,w]][c]; written to both the
//    "embedded" and "embedded_pt" sections (numerically identical forward).
//    grid.x = 2 * EMB_ELEMS/256 = 65536 ; blockIdx.x>>15 selects the input.
// ---------------------------------------------------------------------------
__global__ __launch_bounds__(256)
void vq_gather_kernel(const float* __restrict__ dict, float* __restrict__ out)
{
    const int inp = blockIdx.x >> 15;
    const int o   = ((blockIdx.x & 32767) << 8) + threadIdx.x;   // [0, EMB_ELEMS)
    const float* __restrict__ idx_sec =
        out + (inp ? (4 * EMB_ELEMS + IDX_ELEMS) : (2 * EMB_ELEMS));
    float* __restrict__ emb =
        out + (inp ? (2 * EMB_ELEMS + IDX_ELEMS) : 0);

    const int t  = ((o >> 18) << 10) | (o & 1023);       // n*1024 + p
    const int c  = (o >> 10) & 255;
    const int di = (int)idx_sec[t];
    const float v = dict[di * C_DIM + c];
    emb[o]             = v;                              // embedded
    emb[EMB_ELEMS + o] = v;                              // embedded_pt
}

// ---------------------------------------------------------------------------
extern "C" void kernel_launch(void* const* d_in, const int* in_sizes, int n_in,
                              void* d_out, int out_size, void* d_ws, size_t ws_size,
                              hipStream_t stream)
{
    const float* x0   = (const float*)d_in[0];
    const float* x1   = (const float*)d_in[1];
    const float* dict = (const float*)d_in[2];
    float*  out    = (float*)d_out;
    float*  dnorm  = (float*)d_ws;                       // 1024 f32
    __bf16* dictbf = (__bf16*)(dnorm + D_DIM);           // 1024x256 bf16 (512 KB)

    vq_dictnorm_kernel<<<D_DIM / 256, 256, 0, stream>>>(dict, dnorm);
    vq_dictbf16_kernel<<<(D_DIM * C_DIM / 4) / 256, 256, 0, stream>>>(dict, dictbf);
    vq_argmin_kernel  <<<2 * (IDX_ELEMS / 256), 256, 0, stream>>>(x0, x1, dictbf, dnorm, out);
    vq_gather_kernel  <<<2 * (EMB_ELEMS / 256), 256, 0, stream>>>(dict, out);
}
